// Swinformer_87170656239946
// MI455X (gfx1250) — matrix operand order repaired and checked by reading, not compiled, not run on hardware
//
#include <hip/hip_runtime.h>
#include <hip/hip_bf16.h>
#include <stdint.h>

// ---------------------------------------------------------------- types
typedef __bf16 bf16;
typedef __bf16 v8bf  __attribute__((ext_vector_type(8)));
typedef __bf16 v16bf __attribute__((ext_vector_type(16)));
typedef float  v8f   __attribute__((ext_vector_type(8)));
typedef uint32_t u32x4 __attribute__((ext_vector_type(4)));
typedef int32_t  i32x8 __attribute__((ext_vector_type(8)));
typedef int32_t  i32x4 __attribute__((ext_vector_type(4)));

#define WMMA_BF16(a,b,c) __builtin_amdgcn_wmma_f32_16x16x32_bf16( \
    false, (a), false, (b), (short)0, (c), false, false)

// Model constants: B=32, H=W=32, C=256, CIN=512, L=12, HEADS=8, WS=8, N=64, HD=32, V=1024
#define MTOK   32768                 // B*H*W tokens
#define SCALEQ 0.17677669529663687f  // 32^-0.5

// ---------------------------------------------------------------- helpers
__device__ __forceinline__ bf16 f2bf(float f) {
  union { float f; uint32_t u; } a; a.f = f;
  uint32_t r = (a.u + 0x7FFFu + ((a.u >> 16) & 1u)) >> 16;   // RNE
  union { uint16_t s; bf16 b; } o; o.s = (uint16_t)r;
  return o.b;
}

__device__ __forceinline__ int gid3(int h) {   // Swin shift-mask region id
  return (h < 24) ? 0 : ((h < 28) ? 1 : 2);
}

// A operand, 16x32 bf16: lane half 'hi' holds K=[16*hi, 16*hi+15]
// contiguous (matches B operand order) -> two 16B loads.
__device__ __forceinline__ v16bf load_frag_a(const bf16* row, int hi) {
  v8bf c0 = *(const v8bf*)(row + hi * 16);
  v8bf c1 = *(const v8bf*)(row + hi * 16 + 8);
  return __builtin_shufflevector(c0, c1, 0,1,2,3,4,5,6,7,8,9,10,11,12,13,14,15);
}

// TDM: DMA a [rows x cols] bf16 tile (row stride = srcStride elements) from
// global memory into LDS (row-major, contiguous). 2-D tensor -> groups 2/3 zero.
// D# packing per CDNA5 ISA. 6-arg builtin variant.
__device__ __forceinline__ void tdm_load_tile(const bf16* gsrc, uint32_t lds_addr,
                                              int rows, int cols, int srcStride) {
  const uint64_t ga = (uint64_t)(uintptr_t)gsrc;
  u32x4 g0;
  g0[0] = 1u;                                            // count=1 (valid user D#)
  g0[1] = lds_addr;                                      // lds_addr (bytes)
  g0[2] = (uint32_t)ga;                                  // global_addr[31:0]
  g0[3] = (uint32_t)((ga >> 32) & 0x1FFFFFFu) | (2u << 30); // addr[56:32] | type=2
  i32x8 g1;
  g1[0] = (int)(1u << 16);                               // data_size=1 -> 2 bytes
  g1[1] = (int)((uint32_t)(srcStride & 0xFFFF) << 16);   // tensor_dim0[15:0]
  g1[2] = (int)(((uint32_t)srcStride >> 16) & 0xFFFFu)   // tensor_dim0[31:16]
        | (int)((uint32_t)(rows & 0xFFFF) << 16);        // tensor_dim1[15:0]
  g1[3] = (int)((uint32_t)cols << 16);                   // tile_dim0 = cols
  g1[4] = rows;                                          // tile_dim1 = rows (tile_dim2=0)
  g1[5] = srcStride;                                     // tensor_dim0_stride[31:0]
  g1[6] = 0;                                             // stride hi / dim1_stride
  g1[7] = 0;
  const i32x4 z4 = {0, 0, 0, 0};
  const i32x8 z8 = {0, 0, 0, 0, 0, 0, 0, 0};
  __builtin_amdgcn_tensor_load_to_lds(g0, g1, z4, z4, z8, 0);
}

// ---------------------------------------------------------------- GEMM
// out[m,n] = sum_k A[m,k]*W[n,k]  (A: M x K bf16 row-major, W: N x K bf16 row-major)
// Wave tile: 32(M) x 64(N) = 8 WMMA accumulators; block = 8 waves = 256 x 64.
// The 64x32 weight tile per K-step is block-uniform: staged once into LDS by the
// TDM (double-buffered, TENSORcnt + barrier) instead of 8x redundant L2 fetches.
enum { EPI_CONV = 0, EPI_QKV = 1, EPI_PROJ = 2, EPI_GELU = 3, EPI_RESID = 4, EPI_HEAD = 5 };

template <int EPI>
__global__ __launch_bounds__(256)
void gemm_bf16(const bf16* __restrict__ A, const bf16* __restrict__ Wt,
               const float* __restrict__ bias,
               float* __restrict__ outF, bf16* __restrict__ outB,
               bf16* __restrict__ qb, bf16* __restrict__ kb, bf16* __restrict__ vb,
               int M, int Nn, int K, int shift) {
  __shared__ __align__(64) bf16 BT[2][64][32];   // double-buffered weight tile
  const int lane = threadIdx.x & 31, wv = threadIdx.x >> 5;
  const int lo = lane & 15, hi = lane >> 4;
  const int m0 = (blockIdx.x * 8 + wv) * 32;
  const int n0 = blockIdx.y * 64;
  (void)M;

  v8f acc[2][4] = {};
  const bf16* arow0 = A + (size_t)(m0 + lo) * K;
  const bf16* arow1 = A + (size_t)(m0 + 16 + lo) * K;
  const bf16* wtile = Wt + (size_t)n0 * K;

  if (wv == 0) {                                   // prologue: stage first tile
    tdm_load_tile(wtile, (uint32_t)(uintptr_t)&BT[0][0][0], 64, 32, K);
    __builtin_amdgcn_s_wait_tensorcnt(0);
  }
  __syncthreads();

  int cur = 0;
  for (int k0 = 0; k0 < K; k0 += 32) {
    if (wv == 0 && k0 + 32 < K)                    // async-stage next tile
      tdm_load_tile(wtile + k0 + 32, (uint32_t)(uintptr_t)&BT[cur ^ 1][0][0], 64, 32, K);

    __builtin_prefetch(arow0 + k0 + 64, 0, 3);     // near-scope prefetch on A stream
    v16bf af0 = load_frag_a(arow0 + k0, hi);
    v16bf af1 = load_frag_a(arow1 + k0, hi);
#pragma unroll
    for (int t = 0; t < 4; ++t) {
      // B operand from LDS: lane half 'hi' holds K=[16*hi,16*hi+15] contiguous.
      const v16bf bfv = *(const v16bf*)(&BT[cur][t * 16 + lo][hi * 16]);
      acc[0][t] = WMMA_BF16(af0, bfv, acc[0][t]);
      acc[1][t] = WMMA_BF16(af1, bfv, acc[1][t]);
    }

    if (wv == 0) __builtin_amdgcn_s_wait_tensorcnt(0);
    __syncthreads();
    cur ^= 1;
  }

#pragma unroll
  for (int mi = 0; mi < 2; ++mi) {
#pragma unroll
    for (int t = 0; t < 4; ++t) {
#pragma unroll
      for (int i = 0; i < 8; ++i) {
        const int m = m0 + mi * 16 + i + 8 * hi;  // C/D layout: VGPR i -> M = i + 8*hi
        const int n = n0 + t * 16 + lo;           // lane lo -> N
        float v = acc[mi][t][i];
        if (EPI == EPI_CONV) {
          outF[(size_t)m * Nn + n] = v + bias[n];
        } else if (EPI == EPI_QKV) {
          v += bias[n];
          const int seg = n >> 8, cc = n & 255, hd = cc >> 5, d = cc & 31;
          const int bW = m >> 6, tok = m & 63;
          const size_t idx = ((size_t)(bW * 8 + hd) * 64 + tok) * 32 + d;
          if (seg == 0)      qb[idx] = f2bf(v * SCALEQ);
          else if (seg == 1) kb[idx] = f2bf(v);
          else               vb[idx] = f2bf(v);
        } else if (EPI == EPI_PROJ) {
          // window reverse + roll(+shift) + residual add into t (fp32)
          const int bW = m >> 6, tok = m & 63;
          const int bb = bW >> 4, wy = (bW >> 2) & 3, wx = bW & 3;
          const int r = tok >> 3, c2 = tok & 7;
          const int hh = (wy * 8 + r + shift) & 31;
          const int ww = (wx * 8 + c2 + shift) & 31;
          const size_t p = ((size_t)bb * 32 + hh) * 32 + ww;
          outF[p * 256 + n] += v + bias[n];
        } else if (EPI == EPI_GELU) {
          const float x = v + bias[n];
          const float g = 0.5f * x * (1.0f + erff(x * 0.70710678118654752f));
          outB[(size_t)m * Nn + n] = f2bf(g);
        } else if (EPI == EPI_RESID) {
          outF[(size_t)m * Nn + n] += v + bias[n];
        } else { // EPI_HEAD (no bias)
          outF[(size_t)m * Nn + n] = v;
        }
      }
    }
  }
}

// ---------------------------------------------------------------- LayerNorm (+ shift/window gather) -> bf16
__global__ __launch_bounds__(256)
void ln_kernel(const float* __restrict__ t, const float* __restrict__ g,
               const float* __restrict__ be, bf16* __restrict__ out,
               int windowed, int shift) {
  const int lane = threadIdx.x & 31, wv = threadIdx.x >> 5;
  const int m = blockIdx.x * 8 + wv;
  size_t src;
  if (windowed) {
    const int bW = m >> 6, tok = m & 63;
    const int bb = bW >> 4, wy = (bW >> 2) & 3, wx = bW & 3;
    const int r = tok >> 3, c2 = tok & 7;
    const int hh = (wy * 8 + r + shift) & 31;
    const int ww = (wx * 8 + c2 + shift) & 31;
    src = ((size_t)bb * 32 + hh) * 32 + ww;
  } else {
    src = (size_t)m;
  }
  const float* row = t + src * 256;
  float x[8], s = 0.f;
#pragma unroll
  for (int j = 0; j < 8; ++j) { x[j] = row[lane + j * 32]; s += x[j]; }
#pragma unroll
  for (int o = 16; o; o >>= 1) s += __shfl_xor(s, o, 32);
  const float mean = s * (1.0f / 256.0f);
  float vs = 0.f;
#pragma unroll
  for (int j = 0; j < 8; ++j) { const float d = x[j] - mean; vs += d * d; }
#pragma unroll
  for (int o = 16; o; o >>= 1) vs += __shfl_xor(vs, o, 32);
  const float rstd = rsqrtf(vs * (1.0f / 256.0f) + 1e-5f);
#pragma unroll
  for (int j = 0; j < 8; ++j) {
    const int c = lane + j * 32;
    out[(size_t)m * 256 + c] = f2bf((x[j] - mean) * rstd * g[c] + be[c]);
  }
}

// ---------------------------------------------------------------- attention: one wave per (window, head)
template <int SHIFTED>
__global__ __launch_bounds__(32)
void attn_kernel(const bf16* __restrict__ qb, const bf16* __restrict__ kb,
                 const bf16* __restrict__ vb, const float* __restrict__ rpb,
                 bf16* __restrict__ outB) {
  __shared__ float S[64][64];
  __shared__ bf16  P[64][64];
  __shared__ __align__(64) bf16 VT[32][64];   // v transposed: VT[d][tok]
  __shared__ float BIAS[226];                 // rel-pos bias for this head
  __shared__ int   GID[64];                   // shift-mask region id per token
  const int lane = threadIdx.x, lo = lane & 15, hi = lane >> 4;
  const int pair = blockIdx.x;
  const int bW = pair >> 3, head = pair & 7;
  const int wy = (bW >> 2) & 3, wx = bW & 3;
  const bf16* q = qb + (size_t)pair * 64 * 32;
  const bf16* k = kb + (size_t)pair * 64 * 32;
  const bf16* v = vb + (size_t)pair * 64 * 32;

  // ---- stage per-head bias, mask ids, and v^T into LDS
  for (int idx = lane; idx < 225; idx += 32) BIAS[idx] = rpb[idx * 8 + head];
#pragma unroll
  for (int rr = 0; rr < 2; ++rr) {
    const int tok = lane + rr * 32;
    if (SHIFTED) GID[tok] = gid3(wy * 8 + (tok >> 3)) * 3 + gid3(wx * 8 + (tok & 7));
#pragma unroll
    for (int c4 = 0; c4 < 4; ++c4) {
      v8bf r8 = *(const v8bf*)(v + tok * 32 + c4 * 8);
#pragma unroll
      for (int e = 0; e < 8; ++e) VT[c4 * 8 + e][tok] = r8[e];
    }
  }
  __syncthreads();

  // ---- S = (q*scale) @ k^T, fused +rel-pos-bias (+shift mask)
#pragma unroll
  for (int ti = 0; ti < 4; ++ti) {
    const v16bf af = load_frag_a(q + (ti * 16 + lo) * 32, hi);
    int gm[8];
    if (SHIFTED) {
#pragma unroll
      for (int i = 0; i < 8; ++i) gm[i] = GID[ti * 16 + i + 8 * hi];
    }
#pragma unroll
    for (int tj = 0; tj < 4; ++tj) {
      const v16bf bfv = *(const v16bf*)(k + (tj * 16 + lo) * 32 + hi * 16);
      v8f acc = {};
      acc = WMMA_BF16(af, bfv, acc);
      const int n = tj * 16 + lo;
      const int in_ = n >> 3, jn = n & 7;
      const int gn = SHIFTED ? GID[n] : 0;
#pragma unroll
      for (int i = 0; i < 8; ++i) {
        const int m = ti * 16 + i + 8 * hi;
        const int im = m >> 3, jm = m & 7;
        const int ridx = (im - in_ + 7) * 15 + (jm - jn + 7);
        float val = acc[i] + BIAS[ridx];
        if (SHIFTED) val += (gm[i] != gn) ? -100.0f : 0.0f;
        S[m][n] = val;
      }
    }
  }
  __syncthreads();

  // ---- row softmax (2 rows per lane), result bf16 into P
#pragma unroll
  for (int rr = 0; rr < 2; ++rr) {
    const int m = lane + rr * 32;
    float mx = -1e30f;
    for (int n2 = 0; n2 < 64; ++n2) mx = fmaxf(mx, S[m][n2]);
    float sum = 0.f;
    for (int n2 = 0; n2 < 64; ++n2) { const float e = __expf(S[m][n2] - mx); S[m][n2] = e; sum += e; }
    const float inv = 1.0f / sum;
    for (int n2 = 0; n2 < 64; ++n2) P[m][n2] = f2bf(S[m][n2] * inv);
  }
  __syncthreads();

  // ---- O = P @ v  (B operand from VT in LDS) -> token-major bf16 [bW*64+m, head*32+d]
#pragma unroll
  for (int ti = 0; ti < 4; ++ti) {
    v8f acc[2] = {};
#pragma unroll
    for (int kbk = 0; kbk < 2; ++kbk) {
      const v16bf af = load_frag_a(&P[ti * 16 + lo][kbk * 32], hi);
#pragma unroll
      for (int tj = 0; tj < 2; ++tj) {
        const v16bf bfv = *(const v16bf*)(&VT[tj * 16 + lo][kbk * 32 + hi * 16]);
        acc[tj] = WMMA_BF16(af, bfv, acc[tj]);
      }
    }
#pragma unroll
    for (int tj = 0; tj < 2; ++tj)
#pragma unroll
      for (int i = 0; i < 8; ++i) {
        const int m = ti * 16 + i + 8 * hi;
        const int d = tj * 16 + lo;
        outB[((size_t)bW * 64 + m) * 256 + head * 32 + d] = f2bf(acc[tj][i]);
      }
  }
}

// ---------------------------------------------------------------- misc conversions
__global__ void f2bf_kernel(const float* __restrict__ in, bf16* __restrict__ out, size_t n) {
  const size_t i = (size_t)blockIdx.x * blockDim.x + threadIdx.x;
  if (i < n) out[i] = f2bf(in[i]);
}

// x: [B, CIN, H, W] fp32 -> xb: [B*H*W, CIN] bf16 (token-major)
__global__ void xpose_kernel(const float* __restrict__ x, bf16* __restrict__ xb) {
  const size_t idx = (size_t)blockIdx.x * blockDim.x + threadIdx.x;
  if (idx >= (size_t)MTOK * 512) return;
  const size_t m = idx >> 9;
  const int kk = (int)(idx & 511);
  const size_t b = m >> 10, hw = m & 1023;
  xb[idx] = f2bf(x[(b * 512 + kk) * 1024 + hw]);
}

// ---------------------------------------------------------------- host
extern "C" void kernel_launch(void* const* d_in, const int* in_sizes, int n_in,
                              void* d_out, int out_size, void* d_ws, size_t ws_size,
                              hipStream_t stream) {
  (void)in_sizes; (void)n_in; (void)out_size; (void)ws_size;
  const float* x      = (const float*)d_in[0];
  const float* conv_w = (const float*)d_in[1];
  const float* conv_b = (const float*)d_in[2];
  const float* ln1_g  = (const float*)d_in[3];
  const float* ln1_b  = (const float*)d_in[4];
  const float* qkv_w  = (const float*)d_in[5];
  const float* qkv_b  = (const float*)d_in[6];
  const float* proj_w = (const float*)d_in[7];
  const float* proj_b = (const float*)d_in[8];
  const float* rpb    = (const float*)d_in[9];
  const float* ln2_g  = (const float*)d_in[10];
  const float* ln2_b  = (const float*)d_in[11];
  const float* fc1_w  = (const float*)d_in[12];
  const float* fc1_b  = (const float*)d_in[13];
  const float* fc2_w  = (const float*)d_in[14];
  const float* fc2_b  = (const float*)d_in[15];
  const float* lno_g  = (const float*)d_in[16];
  const float* lno_b  = (const float*)d_in[17];
  const float* head_w = (const float*)d_in[18];

  char* ws = (char*)d_ws;
  size_t off = 0;
  auto take = [&](size_t bytes) -> char* {
    char* p = ws + off; off = (off + bytes + 255) & ~(size_t)255; return p;
  };
  float* t     = (float*)take((size_t)MTOK * 256 * 4);  // residual stream fp32
  bf16*  xw    = (bf16*) take((size_t)MTOK * 256 * 2);  // LN output (GEMM A)
  bf16*  big   = (bf16*) take((size_t)MTOK * 1024 * 2); // conv-in / attn-out / MLP hidden
  bf16*  qbuf  = (bf16*) take((size_t)4096 * 64 * 32 * 2);
  bf16*  kbuf  = (bf16*) take((size_t)4096 * 64 * 32 * 2);
  bf16*  vbuf  = (bf16*) take((size_t)4096 * 64 * 32 * 2);
  bf16*  conv_wb = (bf16*)take((size_t)256 * 512 * 2);
  bf16*  qkv_wb  = (bf16*)take((size_t)12 * 768 * 256 * 2);
  bf16*  proj_wb = (bf16*)take((size_t)12 * 256 * 256 * 2);
  bf16*  fc1_wb  = (bf16*)take((size_t)12 * 1024 * 256 * 2);
  bf16*  fc2_wb  = (bf16*)take((size_t)12 * 256 * 1024 * 2);
  bf16*  head_wb = (bf16*)take((size_t)1024 * 256 * 2);

  auto cvt = [&](const float* src, bf16* dst, size_t n) {
    f2bf_kernel<<<(unsigned)((n + 255) / 256), 256, 0, stream>>>(src, dst, n);
  };
  cvt(conv_w, conv_wb, (size_t)256 * 512);
  cvt(qkv_w,  qkv_wb,  (size_t)12 * 768 * 256);
  cvt(proj_w, proj_wb, (size_t)12 * 256 * 256);
  cvt(fc1_w,  fc1_wb,  (size_t)12 * 1024 * 256);
  cvt(fc2_w,  fc2_wb,  (size_t)12 * 256 * 1024);
  cvt(head_w, head_wb, (size_t)1024 * 256);

  // conv input: transpose x to token-major bf16 (reuses `big`)
  xpose_kernel<<<(unsigned)(((size_t)MTOK * 512 + 255) / 256), 256, 0, stream>>>(x, big);

  // 1x1 conv == GEMM [32768,512]@[256,512]^T + bias -> t
  gemm_bf16<EPI_CONV><<<dim3(MTOK / 256, 256 / 64), 256, 0, stream>>>(
      big, conv_wb, conv_b, t, nullptr, nullptr, nullptr, nullptr, MTOK, 256, 512, 0);

  for (int i = 0; i < 12; ++i) {
    const int shift = (i & 1) ? 4 : 0;
    ln_kernel<<<MTOK / 8, 256, 0, stream>>>(t, ln1_g + i * 256, ln1_b + i * 256, xw, 1, shift);
    gemm_bf16<EPI_QKV><<<dim3(MTOK / 256, 768 / 64), 256, 0, stream>>>(
        xw, qkv_wb + (size_t)i * 768 * 256, qkv_b + i * 768,
        nullptr, nullptr, qbuf, kbuf, vbuf, MTOK, 768, 256, 0);
    if (shift)
      attn_kernel<1><<<4096, 32, 0, stream>>>(qbuf, kbuf, vbuf, rpb + (size_t)i * 225 * 8, big);
    else
      attn_kernel<0><<<4096, 32, 0, stream>>>(qbuf, kbuf, vbuf, rpb + (size_t)i * 225 * 8, big);
    gemm_bf16<EPI_PROJ><<<dim3(MTOK / 256, 256 / 64), 256, 0, stream>>>(
        big, proj_wb + (size_t)i * 256 * 256, proj_b + i * 256,
        t, nullptr, nullptr, nullptr, nullptr, MTOK, 256, 256, shift);
    ln_kernel<<<MTOK / 8, 256, 0, stream>>>(t, ln2_g + i * 256, ln2_b + i * 256, xw, 0, 0);
    gemm_bf16<EPI_GELU><<<dim3(MTOK / 256, 1024 / 64), 256, 0, stream>>>(
        xw, fc1_wb + (size_t)i * 1024 * 256, fc1_b + i * 1024,
        nullptr, big, nullptr, nullptr, nullptr, MTOK, 1024, 256, 0);
    gemm_bf16<EPI_RESID><<<dim3(MTOK / 256, 256 / 64), 256, 0, stream>>>(
        big, fc2_wb + (size_t)i * 256 * 1024, fc2_b + i * 256,
        t, nullptr, nullptr, nullptr, nullptr, MTOK, 256, 1024, 0);
  }

  ln_kernel<<<MTOK / 8, 256, 0, stream>>>(t, lno_g, lno_b, xw, 0, 0);
  gemm_bf16<EPI_HEAD><<<dim3(MTOK / 256, 1024 / 64), 256, 0, stream>>>(
      xw, head_wb, nullptr, (float*)d_out, nullptr, nullptr, nullptr, nullptr,
      MTOK, 1024, 256, 0);
}
